// CustomTransformerEncoder_57423712748007
// MI455X (gfx1250) — matrix operand order repaired and checked by reading, not compiled, not run on hardware
//
#include <hip/hip_runtime.h>

typedef _Float16 half_t;
typedef __attribute__((ext_vector_type(16))) _Float16 v16h;
typedef __attribute__((ext_vector_type(8)))  float    v8f;

#define V_   32000
#define E_   768
#define H_   12
#define L_   6
#define HID_ 3072
#define OUT_ 5
#define B_   8
#define S_   1024
#define D_   64
#define M_   (B_*S_)   // 8192 token rows

// ---------------------------------------------------------------------------
// WMMA fragment loaders (wave32 layouts per CDNA5 ISA 7.12.2)
// ---------------------------------------------------------------------------
union FragU { float4 f[2]; v16h v; };

// A-matrix 16x32 f16: lane row = lane%16, half = lane/16.
// VGPR i<4 : K = k0 + 8*half + {2i,2i+1};  i>=4 : K = k0 + 16 + 8*half + ...
__device__ __forceinline__ v16h ld_frag_a(const half_t* rowp, int k0, int hf) {
  FragU u;
  u.f[0] = *(const float4*)(rowp + k0 + 8*hf);
  u.f[1] = *(const float4*)(rowp + k0 + 16 + 8*hf);
  return u.v;
}

// B-matrix 32x16 f16 (column given per lane): same K order as A-frag:
// elems 0-7 K = k0 + 8*half + 0..7; elems 8-15 K = k0 + 16 + 8*half + 0..7.
__device__ __forceinline__ v16h ld_frag_b(const half_t* colp, int k0, int hf) {
  FragU u;
  u.f[0] = *(const float4*)(colp + k0 + 8*hf);
  u.f[1] = *(const float4*)(colp + k0 + 16 + 8*hf);
  return u.v;
}

__device__ __forceinline__ float wave_sum(float v) {
  v += __shfl_xor(v, 1, 32);
  v += __shfl_xor(v, 2, 32);
  v += __shfl_xor(v, 4, 32);
  v += __shfl_xor(v, 8, 32);
  v += __shfl_xor(v, 16, 32);
  return v;
}

// gfx1250 async copy global->LDS (ASYNCcnt-tracked); lds_off is the raw LDS
// byte offset (= low 32 bits of the generic shared-aperture address).
__device__ __forceinline__ void async_ld_b128(unsigned lds_off, const void* gaddr) {
  asm volatile("global_load_async_to_lds_b128 %0, %1, off"
               :: "v"(lds_off), "v"(gaddr)
               : "memory");
}
__device__ __forceinline__ void wait_async0() {
  asm volatile("s_wait_asynccnt 0x0" ::: "memory");
}

// ---------------------------------------------------------------------------
// Weight prep: f32 [L][E_in][E_out] -> f16 transposed [L*3+which][n][k]
// ---------------------------------------------------------------------------
__global__ __launch_bounds__(256)
void prep_w_kernel(const float* __restrict__ Wq, const float* __restrict__ Wk,
                   const float* __restrict__ Wv, half_t* __restrict__ Wt) {
  size_t i = (size_t)blockIdx.x * 256 + threadIdx.x;
  size_t total = (size_t)3 * L_ * E_ * E_;
  if (i >= total) return;
  size_t k  = i % E_;
  size_t n  = (i / E_) % E_;
  size_t wl = i / ((size_t)E_ * E_);      // layer*3 + which
  int which = (int)(wl % 3);
  int layer = (int)(wl / 3);
  const float* W = which == 0 ? Wq : (which == 1 ? Wk : Wv);
  Wt[i] = (half_t)W[((size_t)layer * E_ + k) * E_ + n];
}

// ---------------------------------------------------------------------------
// Embedding gather + LayerNorm (E=768, 256 thr -> 3 elems/thread)
// ---------------------------------------------------------------------------
__global__ __launch_bounds__(256)
void embed_ln_kernel(const int* __restrict__ idx, const float* __restrict__ tok,
                     const float* __restrict__ pos, const float* __restrict__ g,
                     const float* __restrict__ bb, float* __restrict__ xf,
                     half_t* __restrict__ xh) {
  const int row = blockIdx.x;          // b*S + s
  const int s   = row % S_;
  const int tid = threadIdx.x;
  const int tkn = idx[row];
  __shared__ float sred[8];
  __shared__ float smean, srstd;

  const float* tp = tok + (size_t)tkn * E_;
  const float* pp = pos + (size_t)s * E_;
  float v0 = tp[tid      ] + pp[tid      ];
  float v1 = tp[tid + 256] + pp[tid + 256];
  float v2 = tp[tid + 512] + pp[tid + 512];

  float sm = wave_sum(v0 + v1 + v2);
  if ((tid & 31) == 0) sred[tid >> 5] = sm;
  __syncthreads();
  if (tid == 0) {
    float t = 0;
    for (int i = 0; i < 8; ++i) t += sred[i];
    smean = t * (1.0f / E_);
  }
  __syncthreads();
  float m = smean;
  float d0 = v0 - m, d1 = v1 - m, d2 = v2 - m;
  float sq = wave_sum(d0 * d0 + d1 * d1 + d2 * d2);
  __syncthreads();
  if ((tid & 31) == 0) sred[tid >> 5] = sq;
  __syncthreads();
  if (tid == 0) {
    float t = 0;
    for (int i = 0; i < 8; ++i) t += sred[i];
    srstd = rsqrtf(t * (1.0f / E_) + 1e-5f);
  }
  __syncthreads();
  float rs = srstd;
  float* xo = xf + (size_t)row * E_;
  half_t* ho = xh + (size_t)row * E_;
  float o0 = d0 * rs * g[tid      ] + bb[tid      ];
  float o1 = d1 * rs * g[tid + 256] + bb[tid + 256];
  float o2 = d2 * rs * g[tid + 512] + bb[tid + 512];
  xo[tid] = o0;       ho[tid] = (half_t)o0;
  xo[tid + 256] = o1; ho[tid + 256] = (half_t)o1;
  xo[tid + 512] = o2; ho[tid + 512] = (half_t)o2;
}

// ---------------------------------------------------------------------------
// Per-layer LayerNorm over ctx (f32 in -> f32 + f16 out)
// ---------------------------------------------------------------------------
__global__ __launch_bounds__(256)
void ln_kernel(const float* __restrict__ in, const float* __restrict__ g,
               const float* __restrict__ bb, float* __restrict__ xf,
               half_t* __restrict__ xh) {
  const int row = blockIdx.x;
  const int tid = threadIdx.x;
  __shared__ float sred[8];
  __shared__ float smean, srstd;
  const float* rp = in + (size_t)row * E_;
  float v0 = rp[tid], v1 = rp[tid + 256], v2 = rp[tid + 512];

  float sm = wave_sum(v0 + v1 + v2);
  if ((tid & 31) == 0) sred[tid >> 5] = sm;
  __syncthreads();
  if (tid == 0) {
    float t = 0;
    for (int i = 0; i < 8; ++i) t += sred[i];
    smean = t * (1.0f / E_);
  }
  __syncthreads();
  float m = smean;
  float d0 = v0 - m, d1 = v1 - m, d2 = v2 - m;
  float sq = wave_sum(d0 * d0 + d1 * d1 + d2 * d2);
  __syncthreads();
  if ((tid & 31) == 0) sred[tid >> 5] = sq;
  __syncthreads();
  if (tid == 0) {
    float t = 0;
    for (int i = 0; i < 8; ++i) t += sred[i];
    srstd = rsqrtf(t * (1.0f / E_) + 1e-5f);
  }
  __syncthreads();
  float rs = srstd;
  float* xo = xf + (size_t)row * E_;
  half_t* ho = xh + (size_t)row * E_;
  float o0 = d0 * rs * g[tid      ] + bb[tid      ];
  float o1 = d1 * rs * g[tid + 256] + bb[tid + 256];
  float o2 = d2 * rs * g[tid + 512] + bb[tid + 512];
  xo[tid] = o0;       ho[tid] = (half_t)o0;
  xo[tid + 256] = o1; ho[tid + 256] = (half_t)o1;
  xo[tid + 512] = o2; ho[tid + 512] = (half_t)o2;
}

// ---------------------------------------------------------------------------
// QKV GEMM: Y[M,768] = X[M,768] @ W + bias, f16 in, f32 acc, f16 out.
// grid = (N/64, M/256, 3), block = 256 (8 waves). Wave w -> 32 rows, 64 cols,
// 8 accumulators. k-loop manually unrolled x2 with ping-pong fragment sets:
// parity-p fragments feed 8 WMMAs while parity-(1-p) loads are in flight --
// no register rotation copies, WMMAs never drain loadcnt to zero.
// Weights pre-transposed [n][k] so every fragment load is a contiguous b128.
// ---------------------------------------------------------------------------
__global__ __launch_bounds__(256)
void gemm_qkv_kernel(const half_t* __restrict__ X, const half_t* __restrict__ Wt,
                     const float* __restrict__ bqv, const float* __restrict__ bkv,
                     const float* __restrict__ bvv, half_t* __restrict__ Qo,
                     half_t* __restrict__ Ko, half_t* __restrict__ Vo, int layer) {
  const int which = blockIdx.z;
  const half_t* W = Wt + (size_t)(layer * 3 + which) * E_ * E_;
  const float* bias = (which == 0 ? bqv : (which == 1 ? bkv : bvv)) + (size_t)layer * E_;
  half_t* Y = which == 0 ? Qo : (which == 1 ? Ko : Vo);

  const int n0 = blockIdx.x * 64;
  const int m0 = blockIdx.y * 256;
  const int wave = threadIdx.x >> 5;
  const int lane = threadIdx.x & 31;
  const int nl = lane & 15, hf = lane >> 4;

  const half_t* arow0 = X + (size_t)(m0 + wave * 32 + nl) * E_;
  const half_t* arow1 = arow0 + (size_t)16 * E_;
  const half_t* wrow[4];
#pragma unroll
  for (int t = 0; t < 4; ++t) wrow[t] = W + (size_t)(n0 + t * 16 + nl) * E_;

  v8f acc[2][4] = {};
  v16h Af[2][2];       // [parity][row-block]
  v16h Bf[2][4];       // [parity][n-tile]

  Af[0][0] = ld_frag_a(arow0, 0, hf);
  Af[0][1] = ld_frag_a(arow1, 0, hf);
#pragma unroll
  for (int t = 0; t < 4; ++t) Bf[0][t] = ld_frag_b(wrow[t], 0, hf);

  for (int k0 = 0; k0 < E_; k0 += 64) {
    // load parity 1 (k0+32) while parity 0 computes
    Af[1][0] = ld_frag_a(arow0, k0 + 32, hf);
    Af[1][1] = ld_frag_a(arow1, k0 + 32, hf);
#pragma unroll
    for (int t = 0; t < 4; ++t) Bf[1][t] = ld_frag_b(wrow[t], k0 + 32, hf);
#pragma unroll
    for (int t = 0; t < 4; ++t) {
      acc[0][t] = __builtin_amdgcn_wmma_f32_16x16x32_f16(false, Af[0][0], false, Bf[0][t],
                                                         (short)0, acc[0][t], false, false);
      acc[1][t] = __builtin_amdgcn_wmma_f32_16x16x32_f16(false, Af[0][1], false, Bf[0][t],
                                                         (short)0, acc[1][t], false, false);
    }
    // load parity 0 (k0+64, clamped on last iter) while parity 1 computes
    const int kn = (k0 + 64 < E_) ? k0 + 64 : 0;
    Af[0][0] = ld_frag_a(arow0, kn, hf);
    Af[0][1] = ld_frag_a(arow1, kn, hf);
#pragma unroll
    for (int t = 0; t < 4; ++t) Bf[0][t] = ld_frag_b(wrow[t], kn, hf);
#pragma unroll
    for (int t = 0; t < 4; ++t) {
      acc[0][t] = __builtin_amdgcn_wmma_f32_16x16x32_f16(false, Af[1][0], false, Bf[1][t],
                                                         (short)0, acc[0][t], false, false);
      acc[1][t] = __builtin_amdgcn_wmma_f32_16x16x32_f16(false, Af[1][1], false, Bf[1][t],
                                                         (short)0, acc[1][t], false, false);
    }
  }
#pragma unroll
  for (int t = 0; t < 4; ++t) {
    float bz = bias[n0 + t * 16 + nl];
#pragma unroll
    for (int j = 0; j < 2; ++j)
#pragma unroll
      for (int r = 0; r < 8; ++r) {
        int row = m0 + wave * 32 + j * 16 + r + 8 * hf;
        Y[(size_t)row * E_ + n0 + t * 16 + nl] = (half_t)(acc[j][t][r] + bz);
      }
  }
}

// ---------------------------------------------------------------------------
// Flash attention: grid = (S/64, H, B), block = 128 (4 waves).
// Each wave owns 16 query rows; 32-key blocks; scores via WMMA, online
// softmax through per-wave LDS, P@V via WMMA (P is a native 16x32 A-frag).
// K tile staged with global_load_async_to_lds_b128 (ASYNCcnt path);
// V B-fragments are loaded from LDS BEFORE the softmax so the ds_loads
// overlap the exp/max VALU work; next tile prefetched with global_prefetch.
// ---------------------------------------------------------------------------
__global__ __launch_bounds__(128)
void attn_kernel(const half_t* __restrict__ qb, const half_t* __restrict__ kbuf,
                 const half_t* __restrict__ vbuf, float* __restrict__ ctx) {
  const int qblk = blockIdx.x;
  const int head = blockIdx.y;
  const int bat  = blockIdx.z;
  const int tid  = threadIdx.x;
  const int wave = tid >> 5;
  const int lane = tid & 31;
  const int nl = lane & 15, hf = lane >> 4;

  __shared__ __align__(16) half_t sK[32 * 64];    // [key][d] row-major
  __shared__ __align__(16) half_t sVt[64 * 32];   // [d][key] transposed
  __shared__ __align__(16) half_t sP[4][16 * 32]; // per-wave P tile
  __shared__ float sS[4][16 * 32];                // per-wave scores
  __shared__ float sRow[4][16];                   // per-wave row scales

  const size_t rowbase = (size_t)bat * S_;
  const int qg0 = qblk * 64 + wave * 16;
  const unsigned sKoff = (unsigned)(size_t)(void*)sK;   // raw LDS byte offset

  const half_t* qrow = qb + (rowbase + qg0 + nl) * E_ + head * D_;
  v16h qa0 = ld_frag_a(qrow, 0, hf);
  v16h qa1 = ld_frag_a(qrow, 32, hf);

  v8f acc[4] = {};
  float m_i = -1e30f, l_i = 0.0f;
  const float scale = 0.125f;   // 1/sqrt(64)

  for (int blk = 0; blk < S_ / 32; ++blk) {
    __syncthreads();
    // stage K tile [32][64] via async global->LDS b128 (8 halfs per transfer)
    for (int i = tid; i < 32 * 8; i += 128) {
      int r = i >> 3, c8 = i & 7;
      async_ld_b128(sKoff + (unsigned)i * 16,
                    kbuf + (rowbase + blk * 32 + r) * E_ + head * D_ + c8 * 8);
    }
    // stage V transposed [d][key]: wide global load, scalar LDS scatter
    for (int i = tid; i < 32 * 8; i += 128) {
      int key = i >> 3, c8 = i & 7;
      FragU u;
      u.f[0] = *(const float4*)(vbuf + (rowbase + blk * 32 + key) * E_ + head * D_ + c8 * 8);
#pragma unroll
      for (int j = 0; j < 8; ++j) sVt[(c8 * 8 + j) * 32 + key] = u.v[j];
    }
    // prefetch next block's K/V lines while this block computes
    if (blk + 1 < S_ / 32) {
      int r = tid & 31;
      __builtin_prefetch(kbuf + (rowbase + (blk + 1) * 32 + r) * E_ + head * D_, 0, 0);
      __builtin_prefetch(vbuf + (rowbase + (blk + 1) * 32 + r) * E_ + head * D_, 0, 0);
    }
    wait_async0();
    __syncthreads();

    // scores: S[16q, 32k] = Q[16,64] x K^T  (4 WMMAs)
    v8f sc[2] = {};
#pragma unroll
    for (int tn = 0; tn < 2; ++tn) {
      const half_t* kcol = &sK[(tn * 16 + nl) * 64];
      v16h b0 = ld_frag_b(kcol, 0, hf);
      v16h b1 = ld_frag_b(kcol, 32, hf);
      sc[tn] = __builtin_amdgcn_wmma_f32_16x16x32_f16(false, qa0, false, b0,
                                                      (short)0, sc[tn], false, false);
      sc[tn] = __builtin_amdgcn_wmma_f32_16x16x32_f16(false, qa1, false, b1,
                                                      (short)0, sc[tn], false, false);
    }
    // hoist V B-fragment loads: independent of softmax, overlap its VALU work
    v16h vb[4];
#pragma unroll
    for (int t = 0; t < 4; ++t)
      vb[t] = ld_frag_b(&sVt[(t * 16 + nl) * 32], 0, hf);

#pragma unroll
    for (int tn = 0; tn < 2; ++tn)
#pragma unroll
      for (int r = 0; r < 8; ++r)
        sS[wave][(r + 8 * hf) * 32 + tn * 16 + nl] = sc[tn][r] * scale;
    __syncthreads();

    // online softmax: lanes 0..15 each own one query row
    if (lane < 16) {
      float* srow = &sS[wave][lane * 32];
      float bm = -1e30f;
      for (int j = 0; j < 32; ++j) bm = fmaxf(bm, srow[j]);
      float mnew = fmaxf(m_i, bm);
      float f = __expf(m_i - mnew);
      float bl = 0.0f;
      half_t* prow = &sP[wave][lane * 32];
      for (int j = 0; j < 32; ++j) {
        float p = __expf(srow[j] - mnew);
        prow[j] = (half_t)p;
        bl += p;
      }
      m_i = mnew;
      l_i = l_i * f + bl;
      sRow[wave][lane] = f;
    }
    __syncthreads();

    // rescale accumulator, then ctx[16,64] += P[16,32] @ V[32,64] (4 WMMAs)
    v16h pa = ld_frag_a(&sP[wave][nl * 32], 0, hf);
#pragma unroll
    for (int t = 0; t < 4; ++t) {
#pragma unroll
      for (int r = 0; r < 8; ++r) acc[t][r] *= sRow[wave][r + 8 * hf];
      acc[t] = __builtin_amdgcn_wmma_f32_16x16x32_f16(false, pa, false, vb[t],
                                                      (short)0, acc[t], false, false);
    }
  }

  __syncthreads();
  if (lane < 16) sRow[wave][lane] = 1.0f / l_i;
  __syncthreads();
#pragma unroll
  for (int t = 0; t < 4; ++t)
#pragma unroll
    for (int r = 0; r < 8; ++r) {
      float inv = sRow[wave][r + 8 * hf];
      int q = qg0 + r + 8 * hf;
      ctx[(rowbase + q) * E_ + head * D_ + t * 16 + nl] = acc[t][r] * inv;
    }
}

// ---------------------------------------------------------------------------
// Tail: mean-pool over S, MLP (tiny: 19 MFLOP total, plain VALU f32)
// ---------------------------------------------------------------------------
__global__ __launch_bounds__(256)
void pool_kernel(const float* __restrict__ xf, float* __restrict__ pooled) {
  int i = blockIdx.x * 256 + threadIdx.x;
  if (i >= B_ * E_) return;
  int b = i / E_, e = i % E_;
  float s = 0.0f;
  for (int t = 0; t < S_; ++t) s += xf[((size_t)b * S_ + t) * E_ + e];
  pooled[i] = s * (1.0f / S_);
}

__global__ __launch_bounds__(256)
void mlp1_kernel(const float* __restrict__ pooled, const float* __restrict__ W1,
                 const float* __restrict__ b1, float* __restrict__ h) {
  int i = blockIdx.x * 256 + threadIdx.x;
  if (i >= B_ * HID_) return;
  int b = i / HID_, j = i % HID_;
  float s = b1[j];
  for (int e = 0; e < E_; ++e) s += pooled[b * E_ + e] * W1[(size_t)e * HID_ + j];
  h[i] = fmaxf(s, 0.0f);
}

__global__ __launch_bounds__(64)
void mlp2_kernel(const float* __restrict__ h, const float* __restrict__ W2,
                 const float* __restrict__ b2, float* __restrict__ out) {
  int i = threadIdx.x;
  if (i >= B_ * OUT_) return;
  int b = i / OUT_, o = i % OUT_;
  float s = b2[o];
  for (int j = 0; j < HID_; ++j) s += h[b * HID_ + j] * W2[(size_t)j * OUT_ + o];
  out[i] = s;
}

// ---------------------------------------------------------------------------
extern "C" void kernel_launch(void* const* d_in, const int* in_sizes, int n_in,
                              void* d_out, int out_size, void* d_ws, size_t ws_size,
                              hipStream_t stream) {
  (void)in_sizes; (void)n_in; (void)out_size; (void)ws_size;
  const int*   idx  = (const int*)d_in[0];
  const float* tok  = (const float*)d_in[1];
  const float* pos  = (const float*)d_in[2];
  const float* ln0g = (const float*)d_in[3];
  const float* ln0b = (const float*)d_in[4];
  const float* Wq   = (const float*)d_in[5];
  const float* bq   = (const float*)d_in[6];
  const float* Wk   = (const float*)d_in[7];
  const float* bk   = (const float*)d_in[8];
  const float* Wv   = (const float*)d_in[9];
  const float* bv   = (const float*)d_in[10];
  const float* lng  = (const float*)d_in[11];
  const float* lnb  = (const float*)d_in[12];
  const float* W1   = (const float*)d_in[13];
  const float* b1   = (const float*)d_in[14];
  const float* W2   = (const float*)d_in[15];
  const float* b2   = (const float*)d_in[16];

  char* ws = (char*)d_ws;
  size_t off = 0;
  auto alloc = [&](size_t bytes) -> void* {
    void* p = ws + off;
    off += (bytes + 255) & ~(size_t)255;
    return p;
  };
  half_t* x_h   = (half_t*)alloc((size_t)M_ * E_ * sizeof(half_t));
  float*  x_f   = (float*) alloc((size_t)M_ * E_ * sizeof(float));
  half_t* q_h   = (half_t*)alloc((size_t)M_ * E_ * sizeof(half_t));
  half_t* k_h   = (half_t*)alloc((size_t)M_ * E_ * sizeof(half_t));
  half_t* v_h   = (half_t*)alloc((size_t)M_ * E_ * sizeof(half_t));
  float*  ctx_f = (float*) alloc((size_t)M_ * E_ * sizeof(float));
  half_t* wt    = (half_t*)alloc((size_t)3 * L_ * E_ * E_ * sizeof(half_t));
  float*  pool  = (float*) alloc((size_t)B_ * E_ * sizeof(float));
  float*  hbuf  = (float*) alloc((size_t)B_ * HID_ * sizeof(float));

  {
    size_t total = (size_t)3 * L_ * E_ * E_;
    prep_w_kernel<<<(unsigned)((total + 255) / 256), 256, 0, stream>>>(Wq, Wk, Wv, wt);
  }
  embed_ln_kernel<<<M_, 256, 0, stream>>>(idx, tok, pos, ln0g, ln0b, x_f, x_h);

  for (int l = 0; l < L_; ++l) {
    gemm_qkv_kernel<<<dim3(E_ / 64, M_ / 256, 3), 256, 0, stream>>>(
        x_h, wt, bq, bk, bv, q_h, k_h, v_h, l);
    attn_kernel<<<dim3(S_ / 64, H_, B_), 128, 0, stream>>>(q_h, k_h, v_h, ctx_f);
    ln_kernel<<<M_, 256, 0, stream>>>(ctx_f, lng + (size_t)l * E_, lnb + (size_t)l * E_,
                                      x_f, x_h);
  }

  pool_kernel<<<(B_ * E_ + 255) / 256, 256, 0, stream>>>(x_f, pool);
  mlp1_kernel<<<(B_ * HID_ + 255) / 256, 256, 0, stream>>>(pool, W1, b1, hbuf);
  mlp2_kernel<<<1, 64, 0, stream>>>(hbuf, W2, b2, (float*)d_out);
}
